// DGL_GIN_AttrMasking_62062277427635
// MI455X (gfx1250) — matrix operand order repaired and checked by reading, not compiled, not run on hardware
//
#include <hip/hip_runtime.h>
#include <hip/hip_bf16.h>

// ---------------- problem constants (from reference) ----------------
#define DD   300          // node feature dim
#define HD   600          // hidden dim (2*D)
#define LL   5            // layers
#define KP1  320          // padded K for GEMM1 (D -> 320, multiple of 32)
#define NP1  640          // padded hidden (600 -> 640, multiple of 32)
#define ASTR 328          // LDS row stride for A tile (bank-conflict pad)
#define HSTR 648          // LDS row stride for hidden tile
#define MT   64           // rows per workgroup tile

typedef __attribute__((ext_vector_type(16))) __bf16 v16bf;
typedef __attribute__((ext_vector_type(8)))  __bf16 v8bf;
typedef __attribute__((ext_vector_type(8)))  float  v8f;

static __device__ __forceinline__ v16bf frag_cat(v8bf lo, v8bf hi) {
  return __builtin_shufflevector(lo, hi, 0,1,2,3,4,5,6,7,8,9,10,11,12,13,14,15);
}

static __device__ __forceinline__ v8f wmma_bf16(v16bf a, v16bf b, v8f c) {
  return __builtin_amdgcn_wmma_f32_16x16x32_bf16(false, a, false, b,
                                                 (short)0, c, false, false);
}

// ---------------- weight prep: fp32 -> bf16, transpose + pad ----------------
// W1 [L][300][600] -> W1t [L][640][320]   (W1t[l][n][k] = W1[l][k][n])
__global__ void prep_w1_kernel(const float* __restrict__ W1, __bf16* __restrict__ W1t) {
  const int total = LL * NP1 * KP1;
  for (int i = blockIdx.x * blockDim.x + threadIdx.x; i < total; i += gridDim.x * blockDim.x) {
    int l = i / (NP1 * KP1);
    int r = i - l * (NP1 * KP1);
    int n = r / KP1;
    int k = r - n * KP1;
    float v = (n < HD && k < DD) ? W1[(size_t)l * DD * HD + (size_t)k * HD + n] : 0.f;
    W1t[i] = (__bf16)v;
  }
}
// W2 [L][600][300] -> W2t [L][320][640]
__global__ void prep_w2_kernel(const float* __restrict__ W2, __bf16* __restrict__ W2t) {
  const int total = LL * KP1 * NP1;
  for (int i = blockIdx.x * blockDim.x + threadIdx.x; i < total; i += gridDim.x * blockDim.x) {
    int l = i / (KP1 * NP1);
    int r = i - l * (KP1 * NP1);
    int n = r / NP1;
    int k = r - n * NP1;
    float v = (n < DD && k < HD) ? W2[(size_t)l * HD * DD + (size_t)k * DD + n] : 0.f;
    W2t[i] = (__bf16)v;
  }
}
__global__ void prep_b1_kernel(const float* __restrict__ b1, float* __restrict__ b1p) {
  const int total = LL * NP1;
  for (int i = blockIdx.x * blockDim.x + threadIdx.x; i < total; i += gridDim.x * blockDim.x) {
    int l = i / NP1, n = i - l * NP1;
    b1p[i] = (n < HD) ? b1[l * HD + n] : 0.f;
  }
}

// ---------------- input node embedding ----------------
__global__ void embed_kernel(const int* __restrict__ an, const int* __restrict__ ct,
                             const float* __restrict__ e0, const float* __restrict__ e1,
                             float* __restrict__ h, int n) {
  long long total = (long long)n * DD;
  for (long long i = blockIdx.x * (long long)blockDim.x + threadIdx.x; i < total;
       i += (long long)gridDim.x * blockDim.x) {
    int node = (int)(i / DD);
    int d    = (int)(i - (long long)node * DD);
    h[i] = e0[an[node] * DD + d] + e1[ct[node] * DD + d];
  }
}

// ---------------- message + scatter-sum onto destination nodes ----------------
__global__ void scatter_kernel(const float* __restrict__ h,
                               const int* __restrict__ src, const int* __restrict__ dst,
                               const int* __restrict__ bt, const int* __restrict__ bd,
                               const float* __restrict__ ee0, const float* __restrict__ ee1,
                               float* __restrict__ agg, int ecnt) {
  long long total = (long long)ecnt * DD;
  for (long long i = blockIdx.x * (long long)blockDim.x + threadIdx.x; i < total;
       i += (long long)gridDim.x * blockDim.x) {
    int e = (int)(i / DD);
    int d = (int)(i - (long long)e * DD);
    float v = h[(long long)src[e] * DD + d] + ee0[bt[e] * DD + d] + ee1[bd[e] * DD + d];
    unsafeAtomicAdd(&agg[(long long)dst[e] * DD + d], v);
  }
}

// ---------------- fused GIN MLP: relu(agg@W1+b1)@W2+b2 -> BN (-> ReLU) ----------------
__global__ __launch_bounds__(256, 1)
void gin_mlp_kernel(const float* __restrict__ agg,
                    const __bf16* __restrict__ W1t, const float* __restrict__ b1p,
                    const __bf16* __restrict__ W2t, const float* __restrict__ b2,
                    const float* __restrict__ gamma, const float* __restrict__ beta,
                    const float* __restrict__ mean,  const float* __restrict__ var,
                    float* __restrict__ hout, int nrows, int relu_out) {
  extern __shared__ __bf16 smem[];
  __bf16* A = smem;              // [MT][ASTR]
  __bf16* H = smem + MT * ASTR;  // [MT][HSTR]

  const int tid  = threadIdx.x;
  const int lane = tid & 31;
  const int wave = tid >> 5;
  const int row0 = blockIdx.x * MT;

  // Phase 0: stage agg tile into LDS as bf16 (zero-pad K 300->320, rows >= nrows)
  for (int i = tid; i < MT * KP1; i += 256) {
    int r = i / KP1, c = i - r * KP1;
    int gr = row0 + r;
    float v = (gr < nrows && c < DD) ? agg[(long long)gr * DD + c] : 0.f;
    A[r * ASTR + c] = (__bf16)v;
  }
  __syncthreads();

  // WMMA lane mapping (16-bit A: lane<16 -> kbase 0, else 8; B: kbase 0/8; C/D: rows v + 0/8)
  const int colN  = lane & 15;
  const int kA    = (lane < 16) ? 0 : 8;
  const int kB    = (lane < 16) ? 0 : 8;
  const int mrow  = (wave >> 1) * 16;
  const int mbase = mrow + ((lane < 16) ? 0 : 8);
  const int par   = wave & 1;

  // Phase 1: hidden = relu(A @ W1 + b1) -> bf16 in LDS
  // 4-wide N blocking: one A-fragment load feeds 4 WMMAs (4 independent acc chains)
  for (int j = 0; j < NP1 / 128; ++j) {       // 5 groups
    const int nt0 = j * 8 + par * 4;          // 4 consecutive n-tiles
    v8f acc0 = {}, acc1 = {}, acc2 = {}, acc3 = {};
    const __bf16* arow = A + (mrow + colN) * ASTR + kA;
    const __bf16* brow = W1t + (nt0 * 16 + colN) * KP1 + kB;
#pragma unroll
    for (int kt = 0; kt < KP1 / 32; ++kt) {
      v16bf af = frag_cat(*(const v8bf*)(arow + kt * 32),
                          *(const v8bf*)(arow + kt * 32 + 16));
      v16bf bf0 = frag_cat(*(const v8bf*)(brow + kt * 32), *(const v8bf*)(brow + kt * 32 + 16));
      v16bf bf1 = frag_cat(*(const v8bf*)(brow + 16 * KP1 + kt * 32), *(const v8bf*)(brow + 16 * KP1 + kt * 32 + 16));
      v16bf bf2 = frag_cat(*(const v8bf*)(brow + 32 * KP1 + kt * 32), *(const v8bf*)(brow + 32 * KP1 + kt * 32 + 16));
      v16bf bf3 = frag_cat(*(const v8bf*)(brow + 48 * KP1 + kt * 32), *(const v8bf*)(brow + 48 * KP1 + kt * 32 + 16));
      acc0 = wmma_bf16(af, bf0, acc0);
      acc1 = wmma_bf16(af, bf1, acc1);
      acc2 = wmma_bf16(af, bf2, acc2);
      acc3 = wmma_bf16(af, bf3, acc3);
    }
    v8f accs[4] = {acc0, acc1, acc2, acc3};
#pragma unroll
    for (int t = 0; t < 4; ++t) {
      const int n = (nt0 + t) * 16 + colN;
      const float bias = b1p[n];
#pragma unroll
      for (int v = 0; v < 8; ++v) {
        float x = accs[t][v] + bias;
        H[(mbase + v) * HSTR + n] = (__bf16)(x > 0.f ? x : 0.f);
      }
    }
  }
  __syncthreads();

  // Phase 2: out = BN(H @ W2 + b2) (+ ReLU) -> fp32 global
  // 2-wide N blocking (20 tiles / 2 waves-per-msub -> 5 pairs per wave)
  for (int j = 0; j < KP1 / 64; ++j) {        // 5 groups
    const int nt0 = j * 4 + par * 2;
    v8f acc0 = {}, acc1 = {};
    const __bf16* arow = H + (mrow + colN) * HSTR + kA;
    const __bf16* brow = W2t + (nt0 * 16 + colN) * NP1 + kB;
#pragma unroll 5
    for (int kt = 0; kt < NP1 / 32; ++kt) {
      v16bf af = frag_cat(*(const v8bf*)(arow + kt * 32),
                          *(const v8bf*)(arow + kt * 32 + 16));
      v16bf bf0 = frag_cat(*(const v8bf*)(brow + kt * 32), *(const v8bf*)(brow + kt * 32 + 16));
      v16bf bf1 = frag_cat(*(const v8bf*)(brow + 16 * NP1 + kt * 32), *(const v8bf*)(brow + 16 * NP1 + kt * 32 + 16));
      acc0 = wmma_bf16(af, bf0, acc0);
      acc1 = wmma_bf16(af, bf1, acc1);
    }
    v8f accs[2] = {acc0, acc1};
#pragma unroll
    for (int t = 0; t < 2; ++t) {
      const int n = (nt0 + t) * 16 + colN;
      if (n < DD) {
        const float bias = b2[n];
        const float mu = mean[n];
        const float rs = rsqrtf(var[n] + 1e-5f);
        const float ga = gamma[n];
        const float be = beta[n];
#pragma unroll
        for (int v = 0; v < 8; ++v) {
          int gr = row0 + mbase + v;
          if (gr < nrows) {
            float x = accs[t][v] + bias;
            x = (x - mu) * rs * ga + be;
            if (relu_out) x = x > 0.f ? x : 0.f;
            hout[(long long)gr * DD + n] = x;
          }
        }
      }
    }
  }
}

// ---------------- graph pooling + head ----------------
__global__ void pool_cnt_kernel(const int* __restrict__ gid, float* __restrict__ gcnt, int n) {
  for (int i = blockIdx.x * blockDim.x + threadIdx.x; i < n; i += gridDim.x * blockDim.x)
    unsafeAtomicAdd(&gcnt[gid[i]], 1.0f);
}
__global__ void pool_sum_kernel(const float* __restrict__ h, const int* __restrict__ gid,
                                float* __restrict__ gsum, int n) {
  long long total = (long long)n * DD;
  for (long long i = blockIdx.x * (long long)blockDim.x + threadIdx.x; i < total;
       i += (long long)gridDim.x * blockDim.x) {
    int node = (int)(i / DD);
    int d    = (int)(i - (long long)node * DD);
    unsafeAtomicAdd(&gsum[(long long)gid[node] * DD + d], h[i]);
  }
}
__global__ void pool_avg_kernel(float* __restrict__ gsum, const float* __restrict__ gcnt, int g) {
  int total = g * DD;
  for (int i = blockIdx.x * blockDim.x + threadIdx.x; i < total; i += gridDim.x * blockDim.x) {
    float c = gcnt[i / DD];
    gsum[i] = gsum[i] / fmaxf(c, 1.0f);
  }
}
__global__ void head_kernel(const float* __restrict__ gavg, const float* __restrict__ Wd,
                            const float* __restrict__ bd, float* __restrict__ out) {
  int g = blockIdx.x;
  int o = threadIdx.x;  // 256
  float s = bd[o];
#pragma unroll 4
  for (int k = 0; k < DD; ++k) s += gavg[g * DD + k] * Wd[k * 256 + o];
  out[g * 256 + o] = s;
}

// ---------------- host orchestration ----------------
extern "C" void kernel_launch(void* const* d_in, const int* in_sizes, int n_in,
                              void* d_out, int out_size, void* d_ws, size_t ws_size,
                              hipStream_t stream) {
  const int*   an   = (const int*)d_in[0];
  const int*   ct   = (const int*)d_in[1];
  const int*   bt   = (const int*)d_in[2];
  const int*   bdir = (const int*)d_in[3];
  const int*   src  = (const int*)d_in[4];
  const int*   dst  = (const int*)d_in[5];
  const int*   gid  = (const int*)d_in[6];
  const float* ne0  = (const float*)d_in[8];
  const float* ne1  = (const float*)d_in[9];
  const float* ee0  = (const float*)d_in[10];
  const float* ee1  = (const float*)d_in[11];
  const float* W1   = (const float*)d_in[12];
  const float* b1   = (const float*)d_in[13];
  const float* W2   = (const float*)d_in[14];
  const float* b2   = (const float*)d_in[15];
  const float* bng  = (const float*)d_in[16];
  const float* bnb  = (const float*)d_in[17];
  const float* bnm  = (const float*)d_in[18];
  const float* bnv  = (const float*)d_in[19];
  const float* Wd   = (const float*)d_in[20];
  const float* bd   = (const float*)d_in[21];

  const int N = in_sizes[0];
  const int E = in_sizes[2];
  const int G = out_size / 256;

  // workspace carve-up
  char*  ws  = (char*)d_ws;
  size_t off = 0;
  auto carve = [&](size_t bytes) -> void* {
    void* p = ws + off;
    off += (bytes + 255) & ~(size_t)255;
    return p;
  };
  float*  h    = (float*)carve((size_t)N * DD * sizeof(float));
  float*  agg  = (float*)carve((size_t)N * DD * sizeof(float));
  __bf16* W1t  = (__bf16*)carve((size_t)LL * NP1 * KP1 * sizeof(__bf16));
  __bf16* W2t  = (__bf16*)carve((size_t)LL * KP1 * NP1 * sizeof(__bf16));
  float*  b1p  = (float*)carve((size_t)LL * NP1 * sizeof(float));
  float*  gsum = (float*)carve((size_t)G * DD * sizeof(float));
  float*  gcnt = (float*)carve((size_t)G * sizeof(float));

  const size_t smem = (size_t)(MT * ASTR + MT * HSTR) * sizeof(__bf16);
  (void)hipFuncSetAttribute((const void*)gin_mlp_kernel,
                            hipFuncAttributeMaxDynamicSharedMemorySize, (int)smem);

  prep_w1_kernel<<<512, 256, 0, stream>>>(W1, W1t);
  prep_w2_kernel<<<512, 256, 0, stream>>>(W2, W2t);
  prep_b1_kernel<<<8, 256, 0, stream>>>(b1, b1p);
  embed_kernel<<<4096, 256, 0, stream>>>(an, ct, ne0, ne1, h, N);

  const int mlp_blocks = (N + MT - 1) / MT;
  for (int l = 0; l < LL; ++l) {
    hipMemsetAsync(agg, 0, (size_t)N * DD * sizeof(float), stream);
    scatter_kernel<<<8192, 256, 0, stream>>>(h, src, dst, bt, bdir,
                                             ee0 + (size_t)l * 6 * DD,
                                             ee1 + (size_t)l * 3 * DD, agg, E);
    gin_mlp_kernel<<<mlp_blocks, 256, smem, stream>>>(
        agg,
        W1t + (size_t)l * NP1 * KP1, b1p + (size_t)l * NP1,
        W2t + (size_t)l * KP1 * NP1, b2 + (size_t)l * DD,
        bng + (size_t)l * DD, bnb + (size_t)l * DD,
        bnm + (size_t)l * DD, bnv + (size_t)l * DD,
        h, N, (l < LL - 1) ? 1 : 0);
  }

  hipMemsetAsync(gsum, 0, (size_t)G * DD * sizeof(float), stream);
  hipMemsetAsync(gcnt, 0, (size_t)G * sizeof(float), stream);
  pool_cnt_kernel<<<512, 256, 0, stream>>>(gid, gcnt, N);
  pool_sum_kernel<<<4096, 256, 0, stream>>>(h, gid, gsum, N);
  pool_avg_kernel<<<(G * DD + 255) / 256, 256, 0, stream>>>(gsum, gcnt, G);
  head_kernel<<<G, 256, 0, stream>>>(gsum, Wd, bd, (float*)d_out);
}
